// Qwen3Experts_88630945120690
// MI455X (gfx1250) — compile-verified, matrix operand order checked by reading, not run on hardware
//
#include <hip/hip_runtime.h>
#include <hip/hip_bf16.h>
#include <math.h>

#define DEV __device__ __forceinline__

// Problem dims (fixed by the reference)
constexpr int T    = 4096;
constexpr int H    = 2048;
constexpr int F    = 768;
constexpr int E    = 64;
constexpr int TOPK = 8;
constexpr int TK   = T * TOPK;              // 32768 expanded rows
constexpr int TILE_M    = 128;
constexpr int MAX_TILES = TK / TILE_M + E;  // 320 worst-case row tiles

typedef __attribute__((ext_vector_type(16))) __bf16 v16bf;
typedef __attribute__((ext_vector_type(8)))  float  v8f;

struct alignas(16) bf16x8 { __bf16 v[8]; };

DEV v8f wmma_bf16(v16bf a, v16bf b, v8f c) {
  // D(16x16 f32) = A(16x32 bf16) * B(32x16 bf16) + C
  return __builtin_amdgcn_wmma_f32_16x16x32_bf16(false, a, false, b, (short)0, c,
                                                 false, false);
}

// Build a 16-bit fragment from an LDS row of 32 contiguous bf16 (K dim).
// Lane L -> M (or N) = L%16, group g = L/16; VGPR 0..3 hold K = g*8+[0..7],
// VGPR 4..7 hold K = 16+g*8+[0..7].  Two contiguous 16B runs -> ds_load_b128 x2.
DEV v16bf load_frag(const __bf16* row, int g) {
  v16bf f;
#pragma unroll
  for (int i = 0; i < 8; ++i) {
    f[i]     = row[g * 8 + i];
    f[8 + i] = row[16 + g * 8 + i];
  }
  return f;
}

// Pack two f32 into one bf16 pair (lo -> low halfword): v_cvt_pk_bf16_f32.
DEV unsigned pack_bf16(float lo, float hi) {
  union { __bf16 h[2]; unsigned u; } p;
  p.h[0] = (__bf16)lo;
  p.h[1] = (__bf16)hi;
  return p.u;
}

// Async copy 16B from global (per-lane address) into LDS (per-lane offset).
// Tracked by ASYNCcnt; pair with s_wait_asynccnt before the block barrier.
DEV void async_copy_b128(unsigned lds_byte_off, const void* gaddr) {
  asm volatile("global_load_async_to_lds_b128 %0, %1, off"
               :: "v"(lds_byte_off), "v"(gaddr)
               : "memory");
}
DEV void wait_async() { asm volatile("s_wait_asynccnt 0x0" ::: "memory"); }

// ---------------------------------------------------------------- init
__global__ void k_init(int* __restrict__ counts) {
  if (threadIdx.x < E) counts[threadIdx.x] = 0;
}

// ---------------------------------------------------------------- router top-k
// One wave32 per token; E=64 logits -> 2 per lane. 8 rounds of butterfly argmax
// (tie-break: lowest index, matching jax.lax.top_k), then softmax of the 8.
__global__ __launch_bounds__(256) void k_router(const float* __restrict__ logits,
                                                float* __restrict__ tw,
                                                int* __restrict__ te,
                                                int* __restrict__ counts) {
  const int wave = threadIdx.x >> 5;
  const int lane = threadIdx.x & 31;
  const int t = blockIdx.x * 8 + wave;
  float v0 = logits[t * E + lane];
  float v1 = logits[t * E + 32 + lane];
  const int i0 = lane, i1 = lane + 32;
  float topv[TOPK];
  int   topi[TOPK];
#pragma unroll
  for (int k = 0; k < TOPK; ++k) {
    float v; int idx;
    if (v0 > v1 || (v0 == v1 && i0 < i1)) { v = v0; idx = i0; }
    else                                  { v = v1; idx = i1; }
#pragma unroll
    for (int off = 16; off > 0; off >>= 1) {
      float ov = __shfl_xor(v, off, 32);
      int   oi = __shfl_xor(idx, off, 32);
      if (ov > v || (ov == v && oi < idx)) { v = ov; idx = oi; }
    }
    topv[k] = v; topi[k] = idx;
    if (idx == i0) v0 = -3.4e38f;
    if (idx == i1) v1 = -3.4e38f;
  }
  if (lane == 0) {
    float m = topv[0], s = 0.f, ex[TOPK];
#pragma unroll
    for (int k = 0; k < TOPK; ++k) { ex[k] = __expf(topv[k] - m); s += ex[k]; }
    const float inv = 1.f / s;
#pragma unroll
    for (int k = 0; k < TOPK; ++k) {
      tw[t * TOPK + k] = ex[k] * inv;
      te[t * TOPK + k] = topi[k];
      atomicAdd(&counts[topi[k]], 1);
    }
  }
}

// ---------------------------------------------------------------- scan + tiles
__global__ void k_scan(const int* __restrict__ counts, int* __restrict__ offsets,
                       int* __restrict__ cursor, int* __restrict__ ntiles,
                       int* __restrict__ tExpert, int* __restrict__ tRow,
                       int* __restrict__ tRows) {
  if (threadIdx.x == 0 && blockIdx.x == 0) {
    int off = 0, nt = 0;
    for (int e = 0; e < E; ++e) {
      offsets[e] = off;
      cursor[e]  = off;
      const int c = counts[e];
      for (int r = 0; r < c; r += TILE_M) {
        tExpert[nt] = e;
        tRow[nt]    = off + r;
        tRows[nt]   = (c - r < TILE_M) ? (c - r) : TILE_M;
        ++nt;
      }
      off += c;
    }
    offsets[E] = off;
    ntiles[0]  = nt;
  }
}

// ---------------------------------------------------------------- scatter
__global__ __launch_bounds__(256) void k_scatter(const int* __restrict__ te,
                                                 int* __restrict__ cursor,
                                                 int* __restrict__ slot_token,
                                                 int* __restrict__ slot_of) {
  const int i = blockIdx.x * 256 + threadIdx.x;
  if (i >= TK) return;
  const int e    = te[i];
  const int slot = atomicAdd(&cursor[e], 1);
  slot_token[slot] = i >> 3;  // token = i / TOPK
  slot_of[i]       = slot;
}

// ---------------------------------------------------------------- gather + cvt
// xs[slot][:] = bf16(hidden[token(slot)][:]).  One thread -> 8 elements.
__global__ __launch_bounds__(256) void k_gather(const float* __restrict__ hidden,
                                                const int* __restrict__ slot_token,
                                                __bf16* __restrict__ xs) {
  const unsigned i = blockIdx.x * 256 + threadIdx.x;  // < TK*H/8
  const int row = i >> 8;          // H/8 = 256 chunks per row
  const int c   = (i & 255) << 3;
  const int tok = slot_token[row];
  const float* src = hidden + (size_t)tok * H + c;
  const float4 f0 = *(const float4*)src;
  const float4 f1 = *(const float4*)(src + 4);
  bf16x8 o;
  o.v[0] = (__bf16)f0.x; o.v[1] = (__bf16)f0.y;
  o.v[2] = (__bf16)f0.z; o.v[3] = (__bf16)f0.w;
  o.v[4] = (__bf16)f1.x; o.v[5] = (__bf16)f1.y;
  o.v[6] = (__bf16)f1.z; o.v[7] = (__bf16)f1.w;
  *(bf16x8*)(xs + (size_t)row * H + c) = o;
}

// ---------------------------------------------------------------- GEMM1: gate+up
// Block: 128(M) x 128(N) tile; 8 waves, each owns a 16-row stripe x 8 N-subtiles
// -> 16 WMMAs per wave per K-step.  A tile copied global->LDS with async b128.
// B staged transposed [N][K] via K-pair packing (cvt_pk + ds_store_b32).
__global__ __launch_bounds__(256) void k_gemm_gate_up(
    const __bf16* __restrict__ xs, const float* __restrict__ wg,
    const float* __restrict__ wu, const int* __restrict__ tExpert,
    const int* __restrict__ tRow, const int* __restrict__ tRows,
    const int* __restrict__ ntiles, __bf16* __restrict__ act) {
  const int tile = blockIdx.x;
  if (tile >= ntiles[0]) return;
  const int e     = tExpert[tile];
  const int rs    = tRow[tile];
  const int nrows = tRows[tile];
  const int n0    = blockIdx.y * 128;   // grid.y = F/128 = 6

  __shared__ alignas(16) __bf16 sA[128][32];   // 8 KB
  __shared__ alignas(16) __bf16 sBg[128][32];  // transposed [N][K], 8 KB
  __shared__ alignas(16) __bf16 sBu[128][32];  // 8 KB

  const int tid = threadIdx.x;
  const int lane = tid & 31, w = tid >> 5;
  const int lrow = lane & 15, g = lane >> 4;
  const size_t wbase = (size_t)e * H * F + n0;
  const unsigned sA0 = (unsigned)(size_t)(&sA[0][0]);  // LDS byte offset of sA

  v8f cG[8], cU[8];
#pragma unroll
  for (int s = 0; s < 8; ++s) {
    cG[s] = (v8f){0.f, 0.f, 0.f, 0.f, 0.f, 0.f, 0.f, 0.f};
    cU[s] = (v8f){0.f, 0.f, 0.f, 0.f, 0.f, 0.f, 0.f, 0.f};
  }

  for (int k0 = 0; k0 < H; k0 += 32) {
    // ---- A tile: 128x32 bf16 = 512 x 16B chunks, async global->LDS
#pragma unroll
    for (int q = 0; q < 2; ++q) {
      const int lin = tid + q * 256;      // 0..511
      const int row = lin >> 2;           // 0..127
      const int cb  = (lin & 3) * 16;     // byte column 0..48
      int grow = rs + row;
      if (grow > TK - 1) grow = TK - 1;   // clamp: extra rows computed, not stored
      const void* gp = (const char*)xs + ((size_t)grow * H + k0) * 2 + cb;
      async_copy_b128(sA0 + row * 64 + cb, gp);
    }
    // ---- B tiles: 32(K)x128(N) fp32 each; two K-rows per thread, packed
    //      along K so stores are b32 into the transposed layout.
#pragma unroll
    for (int q = 0; q < 2; ++q) {
      const int lin = tid + q * 256;      // 0..511
      const int kp  = lin >> 5;           // K-pair index 0..15 -> k = 2*kp
      const int n4  = (lin & 31) * 4;     // 0..124
      const float* gp0 = wg + wbase + (size_t)(k0 + 2 * kp) * F + n4;
      const float* up0 = wu + wbase + (size_t)(k0 + 2 * kp) * F + n4;
      const float4 g0 = *(const float4*)gp0;
      const float4 g1 = *(const float4*)(gp0 + F);
      const float4 u0 = *(const float4*)up0;
      const float4 u1 = *(const float4*)(up0 + F);
      *(unsigned*)&sBg[n4 + 0][2 * kp] = pack_bf16(g0.x, g1.x);
      *(unsigned*)&sBg[n4 + 1][2 * kp] = pack_bf16(g0.y, g1.y);
      *(unsigned*)&sBg[n4 + 2][2 * kp] = pack_bf16(g0.z, g1.z);
      *(unsigned*)&sBg[n4 + 3][2 * kp] = pack_bf16(g0.w, g1.w);
      *(unsigned*)&sBu[n4 + 0][2 * kp] = pack_bf16(u0.x, u1.x);
      *(unsigned*)&sBu[n4 + 1][2 * kp] = pack_bf16(u0.y, u1.y);
      *(unsigned*)&sBu[n4 + 2][2 * kp] = pack_bf16(u0.z, u1.z);
      *(unsigned*)&sBu[n4 + 3][2 * kp] = pack_bf16(u0.w, u1.w);
      if (q == 0 && k0 + 32 < H) {
        __builtin_prefetch(gp0 + 32 * F, 0, 1);
        __builtin_prefetch(up0 + 32 * F, 0, 1);
      }
    }
    wait_async();
    __syncthreads();
    const v16bf aF = load_frag(&sA[w * 16 + lrow][0], g);
    // Software-pipelined B fragments: load subtile s+1 before WMMA on s.
    v16bf bG = load_frag(&sBg[lrow][0], g);
    v16bf bU = load_frag(&sBu[lrow][0], g);
#pragma unroll
    for (int s = 0; s < 8; ++s) {
      v16bf nG = bG, nU = bU;
      if (s < 7) {
        nG = load_frag(&sBg[(s + 1) * 16 + lrow][0], g);
        nU = load_frag(&sBu[(s + 1) * 16 + lrow][0], g);
      }
      cG[s] = wmma_bf16(aF, bG, cG[s]);
      cU[s] = wmma_bf16(aF, bU, cU[s]);
      bG = nG; bU = nU;
    }
    __syncthreads();
  }

  // Epilogue: C layout lane -> N=lane%16, VGPR r -> M=r+8*(lane/16)
  const int half = lane >> 4, n = lane & 15;
#pragma unroll
  for (int s = 0; s < 8; ++s) {
#pragma unroll
    for (int r = 0; r < 8; ++r) {
      const int m  = r + half * 8;
      const int lr = w * 16 + m;
      if (lr < nrows) {
        const float gv = cG[s][r], uv = cU[s][r];
        const float sv = gv / (1.0f + __expf(-gv));  // silu
        act[(size_t)(rs + lr) * F + (n0 + s * 16 + n)] = (__bf16)(sv * uv);
      }
    }
  }
}

// ---------------------------------------------------------------- GEMM2: down
// Block: 128(M) x 256(N) tile; 16 subtiles per wave -> 16 WMMAs per K-step.
__global__ __launch_bounds__(256) void k_gemm_down(
    const __bf16* __restrict__ act, const float* __restrict__ wd,
    const int* __restrict__ tExpert, const int* __restrict__ tRow,
    const int* __restrict__ tRows, const int* __restrict__ ntiles,
    float* __restrict__ down) {
  const int tile = blockIdx.x;
  if (tile >= ntiles[0]) return;
  const int e     = tExpert[tile];
  const int rs    = tRow[tile];
  const int nrows = tRows[tile];
  const int n0    = blockIdx.y * 256;   // grid.y = H/256 = 8

  __shared__ alignas(16) __bf16 sA[128][32];   // 8 KB
  __shared__ alignas(16) __bf16 sB[256][32];   // transposed [N][K], 16 KB

  const int tid = threadIdx.x;
  const int lane = tid & 31, w = tid >> 5;
  const int lrow = lane & 15, g = lane >> 4;
  const size_t wbase = (size_t)e * F * H + n0;
  const unsigned sA0 = (unsigned)(size_t)(&sA[0][0]);

  v8f cD[16];
#pragma unroll
  for (int s = 0; s < 16; ++s)
    cD[s] = (v8f){0.f, 0.f, 0.f, 0.f, 0.f, 0.f, 0.f, 0.f};

  for (int k0 = 0; k0 < F; k0 += 32) {
#pragma unroll
    for (int q = 0; q < 2; ++q) {
      const int lin = tid + q * 256;
      const int row = lin >> 2;
      const int cb  = (lin & 3) * 16;
      int grow = rs + row;
      if (grow > TK - 1) grow = TK - 1;
      const void* gp = (const char*)act + ((size_t)grow * F + k0) * 2 + cb;
      async_copy_b128(sA0 + row * 64 + cb, gp);
    }
    // B tile: 32(K)x256(N) fp32, K-pair packed -> b32 stores
#pragma unroll
    for (int q = 0; q < 4; ++q) {
      const int lin = tid + q * 256;      // 0..1023
      const int kp  = lin >> 6;           // 0..15
      const int n4  = (lin & 63) * 4;     // 0..252
      const float* dp0 = wd + wbase + (size_t)(k0 + 2 * kp) * H + n4;
      const float4 d0 = *(const float4*)dp0;
      const float4 d1 = *(const float4*)(dp0 + H);
      *(unsigned*)&sB[n4 + 0][2 * kp] = pack_bf16(d0.x, d1.x);
      *(unsigned*)&sB[n4 + 1][2 * kp] = pack_bf16(d0.y, d1.y);
      *(unsigned*)&sB[n4 + 2][2 * kp] = pack_bf16(d0.z, d1.z);
      *(unsigned*)&sB[n4 + 3][2 * kp] = pack_bf16(d0.w, d1.w);
      if (q == 0 && k0 + 32 < F)
        __builtin_prefetch(dp0 + 32 * H, 0, 1);
    }
    wait_async();
    __syncthreads();
    const v16bf aF = load_frag(&sA[w * 16 + lrow][0], g);
    v16bf bF = load_frag(&sB[lrow][0], g);
#pragma unroll
    for (int s = 0; s < 16; ++s) {
      v16bf nF = bF;
      if (s < 15) nF = load_frag(&sB[(s + 1) * 16 + lrow][0], g);
      cD[s] = wmma_bf16(aF, bF, cD[s]);
      bF = nF;
    }
    __syncthreads();
  }

  const int half = lane >> 4, n = lane & 15;
#pragma unroll
  for (int s = 0; s < 16; ++s) {
#pragma unroll
    for (int r = 0; r < 8; ++r) {
      const int m  = r + half * 8;
      const int lr = w * 16 + m;
      if (lr < nrows)
        down[(size_t)(rs + lr) * H + (n0 + s * 16 + n)] = cD[s][r];
    }
  }
}

// ---------------------------------------------------------------- combine
__global__ __launch_bounds__(256) void k_combine(const float* __restrict__ down,
                                                 const float* __restrict__ tw,
                                                 const int* __restrict__ slot_of,
                                                 float* __restrict__ out) {
  const int t = blockIdx.y;
  const int h = blockIdx.x * 256 + threadIdx.x;
  float acc = 0.f;
#pragma unroll
  for (int k = 0; k < TOPK; ++k) {
    const int i = t * TOPK + k;
    acc += tw[i] * down[(size_t)slot_of[i] * H + h];
  }
  out[(size_t)t * H + h] = acc;
}

// ---------------------------------------------------------------- launch
extern "C" void kernel_launch(void* const* d_in, const int* in_sizes, int n_in,
                              void* d_out, int out_size, void* d_ws, size_t ws_size,
                              hipStream_t stream) {
  const float* hidden = (const float*)d_in[0];  // [T,H]
  const float* logits = (const float*)d_in[1];  // [T,E]
  const float* wg     = (const float*)d_in[2];  // [E,H,F]
  const float* wu     = (const float*)d_in[3];  // [E,H,F]
  const float* wd     = (const float*)d_in[4];  // [E,F,H]
  float* out = (float*)d_out;                   // [T,H] fp32

  // Workspace carve-out (256B aligned slices)
  char* ws = (char*)d_ws;
  size_t o = 0;
  auto take = [&](size_t bytes) -> char* {
    char* p = ws + o;
    o = (o + bytes + 255) & ~(size_t)255;
    return p;
  };
  int*    te         = (int*)take((size_t)TK * 4);
  float*  tw         = (float*)take((size_t)TK * 4);
  int*    slot_token = (int*)take((size_t)TK * 4);
  int*    slot_of    = (int*)take((size_t)TK * 4);
  int*    counts     = (int*)take(E * 4);
  int*    offsets    = (int*)take((E + 1) * 4);
  int*    cursor     = (int*)take(E * 4);
  int*    ntiles     = (int*)take(4);
  int*    tExpert    = (int*)take(MAX_TILES * 4);
  int*    tRow       = (int*)take(MAX_TILES * 4);
  int*    tRows      = (int*)take(MAX_TILES * 4);
  __bf16* xs         = (__bf16*)take((size_t)TK * H * 2);   // 128 MB
  __bf16* act        = (__bf16*)take((size_t)TK * F * 2);   // 48 MB
  float*  down       = (float*)take((size_t)TK * H * 4);    // 256 MB

  k_init<<<1, 64, 0, stream>>>(counts);
  k_router<<<T / 8, 256, 0, stream>>>(logits, tw, te, counts);
  k_scan<<<1, 32, 0, stream>>>(counts, offsets, cursor, ntiles, tExpert, tRow, tRows);
  k_scatter<<<TK / 256, 256, 0, stream>>>(te, cursor, slot_token, slot_of);
  k_gather<<<TK * (H / 8) / 256, 256, 0, stream>>>(hidden, slot_token, xs);
  k_gemm_gate_up<<<dim3(MAX_TILES, F / 128), 256, 0, stream>>>(
      xs, wg, wu, tExpert, tRow, tRows, ntiles, act);
  k_gemm_down<<<dim3(MAX_TILES, H / 256), 256, 0, stream>>>(
      act, wd, tExpert, tRow, tRows, ntiles, down);
  k_combine<<<dim3(H / 256, T), 256, 0, stream>>>(down, tw, slot_of, out);
}
